// Reins_77472620085440
// MI455X (gfx1250) — compile-verified, matrix operand order checked by reading, not run on hardware
//
#include <hip/hip_runtime.h>
#include <hip/hip_bf16.h>

typedef __attribute__((ext_vector_type(16))) __bf16 bf16x16;
typedef __attribute__((ext_vector_type(8)))  __bf16 bf16x8;
typedef __attribute__((ext_vector_type(8)))  float  f32x8;
typedef __attribute__((ext_vector_type(4)))  unsigned u32x4;
typedef __attribute__((ext_vector_type(8)))  int  i32x8;
typedef __attribute__((ext_vector_type(4)))  int  i32x4;

#define WAVE 32

#if defined(__has_builtin)
#  if __has_builtin(__builtin_amdgcn_tensor_load_to_lds) && __has_builtin(__builtin_amdgcn_s_wait_tensorcnt)
#    define HAVE_TDM 1
#  endif
#endif
#ifndef HAVE_TDM
#  define HAVE_TDM 0
#endif

// ---------------------------------------------------------------------------
// bf16 WMMA GEMM with TDM-staged B tile:
//   C[M x N] = A[M x K] * B[N x K]^T (+biasN/+biasM/+addM)
// grid = (Mtiles/8, Ntiles); block = 8 waves; all waves in a block share one
// 16-row B tile (rows nBase..nBase+15, K = Kt*32 elems) staged into LDS by the
// Tensor Data Mover (fallback: cooperative global->LDS copy), then each wave
// computes one 16x16 output tile with v_wmma_f32_16x16x32_bf16, A-fragments
// streamed from global (hot in L2), B-fragments from ds_load_b128.
// Fragment layout per ISA 7.12.2: lane L holds row (L&15); K groups
// {8*(L>>4)..+7} and {16+8*(L>>4)..+7}.
// ---------------------------------------------------------------------------
__global__ void __launch_bounds__(256)
gemm_bf16_wmma(const __bf16* __restrict__ A, int lda,
               const __bf16* __restrict__ B, int ldb,
               float* __restrict__ C, int ldc,
               int M, int N, int Kt,
               const float* __restrict__ biasN,
               const float* __restrict__ biasM,
               const float* __restrict__ addM)
{
    extern __shared__ __bf16 Btile[];           // 16 x (Kt*32) bf16
    const int wave = threadIdx.x >> 5;
    const int lane = threadIdx.x & 31;
    const int K    = Kt * 32;
    const int mBase = (blockIdx.x * 8 + wave) * 16;
    const int nBase = blockIdx.y * 16;

#if HAVE_TDM
    // ---- Tensor Data Mover: stage B tile (16 rows x K) into LDS ----
    if (wave == 0) {
        unsigned long long ga = (unsigned long long)(size_t)(B + (size_t)nBase * ldb);
        unsigned ldsa = (unsigned)(size_t)Btile;           // LDS byte offset = addr[31:0]
        u32x4 g0;
        g0[0] = 1u;                                        // count=1, user descriptor
        g0[1] = ldsa;                                      // lds_addr
        g0[2] = (unsigned)(ga & 0xffffffffull);            // global_addr[31:0]
        g0[3] = (unsigned)((ga >> 32) & 0x1ffffffull)      // global_addr[56:32]
              | 0x80000000u;                               // type=2 ("image")
        i32x8 g1;
        unsigned dim1 = 1u << 20;                          // generous row bound
        g1[0] = (int)(1u << 16);                           // data_size=1 -> 2 bytes
        g1[1] = (int)(((unsigned)K & 0xffffu) << 16);      // tensor_dim0[15:0]
        g1[2] = (int)(((unsigned)K >> 16) | ((dim1 & 0xffffu) << 16)); // dim0 hi | dim1 lo
        g1[3] = (int)((dim1 >> 16) | (((unsigned)K & 0xffffu) << 16)); // dim1 hi | tile_dim0=K
        g1[4] = (int)16;                                   // tile_dim1=16, tile_dim2=0
        g1[5] = (int)(unsigned)ldb;                        // tensor_dim0_stride[31:0]
        g1[6] = 0;                                         // stride0 hi | stride1 lo
        g1[7] = 0;                                         // stride1 hi
        i32x4 gz = {0, 0, 0, 0};
#if defined(__clang_major__) && (__clang_major__ >= 23)
        i32x8 gz8 = {0, 0, 0, 0, 0, 0, 0, 0};
        __builtin_amdgcn_tensor_load_to_lds(g0, g1, gz, gz, gz8, 0);
#else
        __builtin_amdgcn_tensor_load_to_lds(g0, g1, gz, gz, 0);
#endif
        __builtin_amdgcn_s_wait_tensorcnt(0);
    }
#else
    // ---- fallback: cooperative global -> LDS copy ----
    for (int idx = threadIdx.x; idx < 16 * K; idx += 256) {
        int r = idx / K, c = idx - r * K;
        Btile[idx] = B[(size_t)(nBase + r) * ldb + c];
    }
#endif
    __syncthreads();

    const int lo = lane & 15, hi = lane >> 4;
    const __bf16* Arow = A + (size_t)(mBase + lo) * lda;
    const __bf16* Brow = Btile + lo * K;

    f32x8 acc = {};
    for (int kt = 0; kt < Kt; ++kt) {
        int k0 = kt * 32 + 8 * hi;
        int k1 = k0 + 16;
        bf16x8 a0 = *(const bf16x8*)(Arow + k0);
        bf16x8 a1 = *(const bf16x8*)(Arow + k1);
        bf16x8 b0 = *(const bf16x8*)(Brow + k0);
        bf16x8 b1 = *(const bf16x8*)(Brow + k1);
        bf16x16 af, bff;
#pragma unroll
        for (int i = 0; i < 8; ++i) {
            af[i] = a0[i];  af[8 + i] = a1[i];
            bff[i] = b0[i]; bff[8 + i] = b1[i];
        }
        acc = __builtin_amdgcn_wmma_f32_16x16x32_bf16(
            false, af, false, bff, (short)0, acc, false, false);
    }
    // D layout: VGPR r -> row mBase + r + 8*hi, col nBase + lo
#pragma unroll
    for (int r = 0; r < 8; ++r) {
        int row = mBase + r + 8 * hi;
        int col = nBase + lo;
        if (row < M && col < N) {
            float v = acc[r];
            if (biasN) v += biasN[col];
            if (biasM) v += biasM[row];
            if (addM)  v += addM[(size_t)row * ldc + col];
            C[(size_t)row * ldc + col] = v;
        }
    }
}

// ---------------------------------------------------------------------------
// Forward 1024-point radix-2 FFT along C (real input), one (n,b) row per block
// ---------------------------------------------------------------------------
__global__ void __launch_bounds__(256)
fft1024_fwd_real(const float* __restrict__ x,
                 float* __restrict__ oRe, float* __restrict__ oIm)
{
    __shared__ float re[1024], im[1024];
    size_t row = blockIdx.x;
    const float* src = x + row * 1024;
    for (int i = threadIdx.x; i < 1024; i += 256) {
        int r = (int)(__brev((unsigned)i) >> 22);
        re[r] = src[i];
        im[r] = 0.f;
    }
    __syncthreads();
    for (int len = 2; len <= 1024; len <<= 1) {
        int half = len >> 1;
        float ang = -6.28318530717958647692f / (float)len;
        for (int j = threadIdx.x; j < 512; j += 256) {
            int blk = j / half, pos = j - blk * half;
            int i0 = blk * len + pos, i1 = i0 + half;
            float s, c; __sincosf(ang * (float)pos, &s, &c);
            float tr = c * re[i1] - s * im[i1];
            float ti = c * im[i1] + s * re[i1];
            re[i1] = re[i0] - tr; im[i1] = im[i0] - ti;
            re[i0] += tr;         im[i0] += ti;
        }
        __syncthreads();
    }
    for (int i = threadIdx.x; i < 1024; i += 256) {
        oRe[row * 1024 + i] = re[i];
        oIm[row * 1024 + i] = im[i];
    }
}

// ---------------------------------------------------------------------------
// Direct 8-point DFT along B; sign=-1 forward, sign=+1 inverse; scaled
// ---------------------------------------------------------------------------
__constant__ float C8t[8] = { 1.f, 0.70710678f, 0.f, -0.70710678f, -1.f, -0.70710678f, 0.f, 0.70710678f };
__constant__ float S8t[8] = { 0.f, 0.70710678f, 1.f, 0.70710678f, 0.f, -0.70710678f, -1.f, -0.70710678f };

__global__ void __launch_bounds__(256)
dft8_along_b(float* __restrict__ re, float* __restrict__ im,
             int ncols, float sign, float scale)
{
    int idx = blockIdx.x * 256 + threadIdx.x;
    int total = 1024 * ncols;
    if (idx >= total) return;
    int nn = idx / ncols, cc = idx - nn * ncols;
    size_t base = (size_t)nn * 8192 + cc;
    float xr[8], xi[8], yr[8], yi[8];
#pragma unroll
    for (int b = 0; b < 8; ++b) { xr[b] = re[base + b * 1024]; xi[b] = im[base + b * 1024]; }
#pragma unroll
    for (int k = 0; k < 8; ++k) {
        float sr = 0.f, si = 0.f;
#pragma unroll
        for (int b = 0; b < 8; ++b) {
            int t = (k * b) & 7;
            float c = C8t[t], s = sign * S8t[t];
            sr += xr[b] * c - xi[b] * s;
            si += xr[b] * s + xi[b] * c;
        }
        yr[k] = sr * scale; yi[k] = si * scale;
    }
#pragma unroll
    for (int k = 0; k < 8; ++k) { re[base + k * 1024] = yr[k]; im[base + k * 1024] = yi[k]; }
}

// ---------------------------------------------------------------------------
// amp/phase with denormal replacement; also emit bf16 copies for WMMA
// ---------------------------------------------------------------------------
__global__ void __launch_bounds__(256)
amp_phase_kernel(const float* __restrict__ re, const float* __restrict__ im,
                 float* __restrict__ amp, float* __restrict__ phase,
                 __bf16* __restrict__ amp_b, __bf16* __restrict__ phase_b, int n)
{
    int i = blockIdx.x * 256 + threadIdx.x;
    if (i >= n) return;
    float r = re[i], m = im[i];
    float p = r * r + m * m;
    p = (p < 1e-5f && p > -1e-5f) ? 1e-5f : p;
    float a = sqrtf(p);
    float rr = (r < 1e-5f && r > -1e-5f) ? 1e-5f : r;
    float ph = atan2f(m, rr);
    amp[i] = a; phase[i] = ph;
    amp_b[i] = (__bf16)a; phase_b[i] = (__bf16)ph;
}

// ---------------------------------------------------------------------------
// Conversions
// ---------------------------------------------------------------------------
__global__ void __launch_bounds__(256)
f32_to_bf16_kernel(const float* __restrict__ src, __bf16* __restrict__ dst, int n)
{
    int i = blockIdx.x * 256 + threadIdx.x;
    if (i < n) dst[i] = (__bf16)src[i];
}

__global__ void __launch_bounds__(256)
tokens_to_bf16(const float* __restrict__ toks, const int* __restrict__ layer,
               __bf16* __restrict__ dst)   // dst: 128 x 1024, zero-padded rows
{
    int i = blockIdx.x * 256 + threadIdx.x;
    if (i >= 128 * 1024) return;
    int row = i >> 10;
    dst[i] = (row < 100) ? (__bf16)toks[(size_t)layer[0] * 102400 + i] : (__bf16)0.f;
}

__global__ void __launch_bounds__(256)
vT_to_bf16(const float* __restrict__ vT, __bf16* __restrict__ dst) // 1024 x 128
{
    int i = blockIdx.x * 256 + threadIdx.x;
    if (i >= 1024 * 128) return;
    int c = i >> 7, j = i & 127;
    dst[i] = (j < 99) ? (__bf16)vT[(size_t)c * 112 + j] : (__bf16)0.f;
}

// ---------------------------------------------------------------------------
// Optional instance-norm + softmax over m (100), one wave per row;
// writes bf16 attn[:,1:] into zero-padded 128-wide buffer (K-pad for WMMA)
// ---------------------------------------------------------------------------
__global__ void __launch_bounds__(256)
norm_softmax_bf16(const float* __restrict__ attn, int lda,
                  __bf16* __restrict__ outb, int ldb, int rows, int doNorm)
{
    int wave = threadIdx.x >> 5, lane = threadIdx.x & 31;
    int row = blockIdx.x * 8 + wave;
    if (row >= rows) return;
    const float* src = attn + (size_t)row * lda;
    float v[4];
#pragma unroll
    for (int i = 0; i < 4; ++i) { int m = lane + 32 * i; v[i] = (m < 100) ? src[m] : 0.f; }
    if (doNorm) {
        float s = v[0] + v[1] + v[2] + v[3];
        for (int off = 16; off; off >>= 1) s += __shfl_xor(s, off, WAVE);
        float mu = s * 0.01f;
        float q = 0.f;
#pragma unroll
        for (int i = 0; i < 4; ++i) { int m = lane + 32 * i; if (m < 100) { float d = v[i] - mu; q += d * d; } }
        for (int off = 16; off; off >>= 1) q += __shfl_xor(q, off, WAVE);
        float inv = rsqrtf(q * 0.01f + 1e-5f);
#pragma unroll
        for (int i = 0; i < 4; ++i) v[i] = (v[i] - mu) * inv;
    }
    float mx = -3.4e38f;
#pragma unroll
    for (int i = 0; i < 4; ++i) { int m = lane + 32 * i; if (m < 100) mx = fmaxf(mx, v[i] * 0.03125f); }
    for (int off = 16; off; off >>= 1) mx = fmaxf(mx, __shfl_xor(mx, off, WAVE));
    float se = 0.f;
#pragma unroll
    for (int i = 0; i < 4; ++i) {
        int m = lane + 32 * i;
        if (m < 100) { v[i] = __expf(v[i] * 0.03125f - mx); se += v[i]; } else v[i] = 0.f;
    }
    for (int off = 16; off; off >>= 1) se += __shfl_xor(se, off, WAVE);
    float inv = 1.0f / se;
    __bf16* dst = outb + (size_t)row * ldb;
#pragma unroll
    for (int i = 0; i < 4; ++i) {
        int m = lane + 32 * i;
        if (m >= 1 && m < 100) dst[m - 1] = (__bf16)(v[i] * inv);
    }
    if (lane < 29) dst[99 + lane] = (__bf16)0.f;   // zero K-pad cols 99..127
}

// ---------------------------------------------------------------------------
// z = (cos(dP) + i sin(dP)) * dA / sqrt(2C)
// ---------------------------------------------------------------------------
__global__ void __launch_bounds__(256)
build_z(const float* __restrict__ dP, const float* __restrict__ dA,
        float* __restrict__ zRe, float* __restrict__ zIm, int n)
{
    int i = blockIdx.x * 256 + threadIdx.x;
    if (i >= n) return;
    float s, c; __sincosf(dP[i], &s, &c);
    float a = dA[i] * 0.02209708691f;   // 1/sqrt(2048)
    zRe[i] = c * a; zIm[i] = s * a;
}

// ---------------------------------------------------------------------------
// Hermitian-extend 513 cols -> 1024-pt inverse FFT (ortho part 1/sqrt(1024)),
// fused with out = x + delta*scale
// ---------------------------------------------------------------------------
__global__ void __launch_bounds__(256)
irfft1024_store(const float* __restrict__ zRe, const float* __restrict__ zIm,
                const float* __restrict__ feats, const float* __restrict__ scale_p,
                float* __restrict__ out)
{
    __shared__ float re[1024], im[1024];
    size_t row = blockIdx.x;
    const float* sre = zRe + row * 1024;
    const float* sim = zIm + row * 1024;
    for (int i = threadIdx.x; i < 1024; i += 256) {
        int r = (int)(__brev((unsigned)i) >> 22);
        float vr, vi;
        if (i <= 512) { vr = sre[i]; vi = sim[i]; }
        else          { vr = sre[1024 - i]; vi = -sim[1024 - i]; }
        re[r] = vr; im[r] = vi;
    }
    __syncthreads();
    for (int len = 2; len <= 1024; len <<= 1) {
        int half = len >> 1;
        float ang = 6.28318530717958647692f / (float)len;   // inverse: +i
        for (int j = threadIdx.x; j < 512; j += 256) {
            int blk = j / half, pos = j - blk * half;
            int i0 = blk * len + pos, i1 = i0 + half;
            float s, c; __sincosf(ang * (float)pos, &s, &c);
            float tr = c * re[i1] - s * im[i1];
            float ti = c * im[i1] + s * re[i1];
            re[i1] = re[i0] - tr; im[i1] = im[i0] - ti;
            re[i0] += tr;         im[i0] += ti;
        }
        __syncthreads();
    }
    float sc = scale_p[0] * 0.03125f;   // scale * 1/sqrt(1024)
    const float* xr = feats + 8192 + row * 1024;
    float* orow = out + 8192 + row * 1024;
    for (int i = threadIdx.x; i < 1024; i += 256)
        orow[i] = xr[i] + re[i] * sc;
}

__global__ void __launch_bounds__(256)
copy_cls(const float* __restrict__ feats, float* __restrict__ out)
{
    int i = blockIdx.x * 256 + threadIdx.x;
    if (i < 8192) out[i] = feats[i];
}

// ---------------------------------------------------------------------------
extern "C" void kernel_launch(void* const* d_in, const int* in_sizes, int n_in,
                              void* d_out, int out_size, void* d_ws, size_t ws_size,
                              hipStream_t stream)
{
    const float* feats   = (const float*)d_in[0];
    const float* toks1   = (const float*)d_in[1];
    const float* toks2   = (const float*)d_in[2];
    const float* w_t2f   = (const float*)d_in[3];
    const float* b_t2f   = (const float*)d_in[4];
    const float* w_df    = (const float*)d_in[5];
    const float* b_df    = (const float*)d_in[6];
    const float* w_t2f2  = (const float*)d_in[7];
    const float* b_t2f2  = (const float*)d_in[8];
    const float* w_df2   = (const float*)d_in[9];
    const float* b_df2   = (const float*)d_in[10];
    const float* scale_p = (const float*)d_in[11];
    const int*   layer_p = (const int*)d_in[12];
    float* out = (float*)d_out;

    const size_t NE = 8192ull * 1024ull;

    // workspace arena
    char* w = (char*)d_ws;
    size_t off = 0;
    auto alloc = [&](size_t bytes) -> void* {
        void* p = w + off; off += (bytes + 255) & ~(size_t)255; return p;
    };
    float*  fftRe   = (float*) alloc(NE * 4);
    float*  fftIm   = (float*) alloc(NE * 4);
    float*  phase   = (float*) alloc(NE * 4);
    float*  amp     = (float*) alloc(NE * 4);
    __bf16* phase_b = (__bf16*)alloc(NE * 2);
    __bf16* amp_b   = (__bf16*)alloc(NE * 2);
    __bf16* s_b     = (__bf16*)alloc(NE * 2);
    __bf16* w1_b    = (__bf16*)alloc(1048576ull * 2);
    __bf16* w2_b    = (__bf16*)alloc(1048576ull * 2);
    __bf16* w3_b    = (__bf16*)alloc(1048576ull * 2);
    __bf16* w4_b    = (__bf16*)alloc(1048576ull * 2);
    __bf16* t1_b    = (__bf16*)alloc(128ull * 1024 * 2);
    __bf16* t2_b    = (__bf16*)alloc(128ull * 1024 * 2);
    float*  attnF   = (float*) alloc(8192ull * 112 * 4);
    __bf16* attnS_b = (__bf16*)alloc(8192ull * 128 * 2);
    float*  vTf     = (float*) alloc(1024ull * 112 * 4);
    __bf16* vT_b    = (__bf16*)alloc(1024ull * 128 * 2);

    // ---- FFT2 ----
    fft1024_fwd_real<<<8192, 256, 0, stream>>>(feats + 8192, fftRe, fftIm);
    dft8_along_b<<<(1024 * 1024 + 255) / 256, 256, 0, stream>>>(fftRe, fftIm, 1024, -1.f, 1.f);
    amp_phase_kernel<<<(int)((NE + 255) / 256), 256, 0, stream>>>(
        fftRe, fftIm, amp, phase, amp_b, phase_b, (int)NE);

    // ---- weight / token conversions ----
    f32_to_bf16_kernel<<<4096, 256, 0, stream>>>(w_t2f,  w1_b, 1048576);
    f32_to_bf16_kernel<<<4096, 256, 0, stream>>>(w_df,   w2_b, 1048576);
    f32_to_bf16_kernel<<<4096, 256, 0, stream>>>(w_t2f2, w3_b, 1048576);
    f32_to_bf16_kernel<<<4096, 256, 0, stream>>>(w_df2,  w4_b, 1048576);
    tokens_to_bf16<<<512, 256, 0, stream>>>(toks1, layer_p, t1_b);
    tokens_to_bf16<<<512, 256, 0, stream>>>(toks2, layer_p, t2_b);

    const size_t sh1024 = 16ull * 1024 * 2;   // B tile LDS bytes for K=1024
    const size_t sh128  = 16ull * 128 * 2;    // B tile LDS bytes for K=128

    // ---- two delta_feat pipelines (phase: no norm; amp: instance-norm) ----
    for (int pipe = 0; pipe < 2; ++pipe) {
        const __bf16* Xb   = pipe ? amp_b  : phase_b;
        float*        Xf   = pipe ? amp    : phase;
        const __bf16* Tb   = pipe ? t2_b   : t1_b;
        const __bf16* Wt2f = pipe ? w3_b   : w1_b;
        const float*  Bt2f = pipe ? b_t2f2 : b_t2f;
        const __bf16* Wdf  = pipe ? w4_b   : w2_b;
        const float*  Bdf  = pipe ? b_df2  : b_df;

        // attn = X @ T^T : M=8192 N=100 K=1024
        gemm_bf16_wmma<<<dim3(64, 7), 256, sh1024, stream>>>(
            Xb, 1024, Tb, 1024, attnF, 112, 8192, 100, 32, nullptr, nullptr, nullptr);
        // (instance-norm) + softmax -> bf16 padded (8192 x 128)
        norm_softmax_bf16<<<1024, 256, 0, stream>>>(attnF, 112, attnS_b, 128, 8192, pipe);
        // vT[c][m] = sum_k W[c][k]*T[m+1][k] + b[c] : M=1024 N=99 K=1024 (row bias)
        gemm_bf16_wmma<<<dim3(8, 7), 256, sh1024, stream>>>(
            Wt2f, 1024, Tb + 1024, 1024, vTf, 112, 1024, 99, 32, nullptr, Bt2f, nullptr);
        vT_to_bf16<<<512, 256, 0, stream>>>(vTf, vT_b);
        // df + X : M=8192 N=1024 K=128 (zero-padded), addM = X
        gemm_bf16_wmma<<<dim3(64, 64), 256, sh128, stream>>>(
            attnS_b, 128, vT_b, 128, fftRe, 1024, 8192, 1024, 4, nullptr, nullptr, Xf);
        f32_to_bf16_kernel<<<(int)((NE + 255) / 256), 256, 0, stream>>>(fftRe, s_b, (int)NE);
        // d = s @ Wdf^T + b : M=8192 N=1024 K=1024 (col bias), overwrite Xf
        gemm_bf16_wmma<<<dim3(64, 64), 256, sh1024, stream>>>(
            s_b, 1024, Wdf, 1024, Xf, 1024, 8192, 1024, 32, Bdf, nullptr, nullptr);
    }

    // ---- z build + irfft2 (ortho) + residual ----
    build_z<<<(int)((NE + 255) / 256), 256, 0, stream>>>(phase, amp, fftRe, fftIm, (int)NE);
    dft8_along_b<<<(1024 * 513 + 255) / 256, 256, 0, stream>>>(fftRe, fftIm, 513, 1.f, 0.35355339059f);
    irfft1024_store<<<8192, 256, 0, stream>>>(fftRe, fftIm, feats, scale_p, out);
    copy_cls<<<32, 256, 0, stream>>>(feats, out);
}
